// PtrDecoder_42245298323489
// MI455X (gfx1250) — compile-verified, matrix operand order checked by reading, not run on hardware
//
#include <hip/hip_runtime.h>

#define KW   24
#define BSZ  64
#define PLY  32
#define EMB  512
#define BTOT (KW * BSZ)   // 1536 rows

// ---- types -----------------------------------------------------------------
typedef __attribute__((ext_vector_type(16))) __bf16 v16bf;
typedef __attribute__((ext_vector_type(8)))  float  v8f;

// bf16 storage type that is safe in host-visible signatures
typedef unsigned short bf16s;

__device__ __forceinline__ bf16s f2bf(float f) {
    unsigned u = __builtin_bit_cast(unsigned, f);
    return (bf16s)((u + 0x7FFFu + ((u >> 16) & 1u)) >> 16);
}

__device__ __forceinline__ v8f wmma_bf16(v16bf a, v16bf b, v8f c) {
    // D = A(16x32 bf16) x B(32x16 bf16) + C(16x16 f32)
    return __builtin_amdgcn_wmma_f32_16x16x32_bf16(false, a, false, b,
                                                   (short)0, c, false, false);
}

// Direct 32-byte vector load of a WMMA operand (no element-wise glue).
// A operand: lane L -> row M=L%16, 16 contiguous K at kc+(L/16)*16.
__device__ __forceinline__ v16bf load_a16(const bf16s* base, int ldk, int lane, int kc) {
    int row = lane & 15, half = (lane >> 4) & 1;
    return *(const v16bf*)(base + row * ldk + kc + half * 16);
}

// B operand for D=A*W^T: lane L -> output col j=col0+L%16, 16 contiguous K
// at kc+(L/16)*16 from row-major W[j][k].
__device__ __forceinline__ v16bf load_b16(const bf16s* w, int ldk, int lane, int col0, int kc) {
    int col = col0 + (lane & 15), half = (lane >> 4) & 1;
    return *(const v16bf*)(w + (size_t)col * ldk + kc + half * 16);
}

// ---- K0: one-time weight conversion to bf16, bias fusion -------------------
__global__ void __launch_bounds__(256)
k_prep(const float* __restrict__ W_ih, const float* __restrict__ W_hh,
       const float* __restrict__ b_ih, const float* __restrict__ b_hh,
       const float* __restrict__ Wq,   const float* __restrict__ Wc,
       bf16s* __restrict__ Wcat, bf16s* __restrict__ Wqb,
       bf16s* __restrict__ Wcb,  float* __restrict__ bcomb) {
    const int NCAT = 2048 * 1024;
    const int NQ   = 512 * 512;
    const int total = NCAT + 2 * NQ + 2048;
    for (int i = blockIdx.x * blockDim.x + threadIdx.x; i < total;
         i += gridDim.x * blockDim.x) {
        if (i < NCAT) {
            int j = i >> 10, k = i & 1023;
            float x = (k < 512) ? W_ih[j * 512 + k] : W_hh[j * 512 + (k - 512)];
            Wcat[i] = f2bf(x);
        } else if (i < NCAT + NQ) {
            int o = i - NCAT; Wqb[o] = f2bf(Wq[o]);
        } else if (i < NCAT + 2 * NQ) {
            int o = i - NCAT - NQ; Wcb[o] = f2bf(Wc[o]);
        } else {
            int j = i - NCAT - 2 * NQ; bcomb[j] = b_ih[j] + b_hh[j];
        }
    }
}

// ---- K1: gather decoder inputs, h0=c0=sum_p enc ---------------------------
// enc(B,ply,e) row b maps to enc4[(b%KW)*BSZ + b/KW] (transpose (1,0,2,3)).
__global__ void __launch_bounds__(256)
k_gather(const float* __restrict__ enc4, const int* __restrict__ idx,
         const float* __restrict__ init_embed,
         bf16s* __restrict__ steps, float* __restrict__ cbuf,
         bf16s* __restrict__ hbf) {
    int b = blockIdx.x, tid = threadIdx.x;
    const float* encb = enc4 + (size_t)((b % KW) * BSZ + (b / KW)) * (PLY * EMB);
    for (int e = tid; e < EMB; e += blockDim.x) {
        float s = 0.f;
        for (int p = 0; p < PLY; ++p) s += encb[p * EMB + e];
        cbuf[(size_t)b * EMB + e] = s;
        hbf[(size_t)b * EMB + e]  = f2bf(s);
        steps[(size_t)b * PLY * EMB + e] = f2bf(init_embed[e]);   // step 0
    }
    for (int t = 1; t < PLY; ++t) {
        int src = idx[b * PLY + (t - 1)];
        for (int e = tid; e < EMB; e += blockDim.x)
            steps[(size_t)b * PLY * EMB + (size_t)t * EMB + e] = f2bf(encb[src * EMB + e]);
    }
}

// ---- K2: ctx = enc @ Wc^T  (B*PLY x 512, K=512) via WMMA -------------------
__global__ void __launch_bounds__(256)
k_ctx(const float* __restrict__ enc4, const bf16s* __restrict__ Wcb,
      float* __restrict__ ctx) {
    __shared__ __align__(32) bf16s sA[16 * 512];
    int r0 = blockIdx.x * 16;
    int tid = threadIdx.x, lane = tid & 31, wv = tid >> 5;
    for (int i = tid; i < 16 * 512; i += blockDim.x) {
        int r = i >> 9, e = i & 511;
        int row = r0 + r, b = row >> 5, p = row & 31;
        const float* encr = enc4 + (size_t)((b % KW) * BSZ + b / KW) * (PLY * EMB) + p * EMB;
        sA[i] = f2bf(encr[e]);
    }
    __syncthreads();
    v8f acc[4] = {};
    int jt0 = wv * 4;                              // 8 waves * 4 tiles = 32 col tiles
    for (int kc = 0; kc < 512; kc += 32) {
        v16bf a = load_a16(sA, 512, lane, kc);
#pragma unroll
        for (int ii = 0; ii < 4; ++ii)
            acc[ii] = wmma_bf16(a, load_b16(Wcb, 512, lane, (jt0 + ii) * 16, kc), acc[ii]);
    }
    int half = lane >> 4, N = lane & 15;
#pragma unroll
    for (int ii = 0; ii < 4; ++ii) {
        int col = (jt0 + ii) * 16 + N;
#pragma unroll
        for (int k = 0; k < 8; ++k)
            ctx[(size_t)(r0 + k + 8 * half) * EMB + col] = acc[ii][k];
    }
}

// ---- K3: one fused LSTM + attention step for 16 rows -----------------------
__global__ void __launch_bounds__(256)
k_step(int t, const bf16s* __restrict__ steps, const bf16s* __restrict__ Wcat,
       const float* __restrict__ bcomb, const bf16s* __restrict__ Wqb,
       const float* __restrict__ bq, const float* __restrict__ ctx,
       const float* __restrict__ vvec, float* __restrict__ cbuf,
       bf16s* __restrict__ hbf, float* __restrict__ out_att,
       float* __restrict__ out_arg) {
    __shared__ __align__(32) bf16s sA[16 * 1024];  // 32KB; later reused as f32 q-buffer
    __shared__ __align__(32) bf16s sH[16 * 512];   // 16KB new-h in bf16
    __shared__ float sSc[16 * 32];
    __shared__ float sMl[16];
    int r0 = blockIdx.x * 16;
    int tid = threadIdx.x, lane = tid & 31, wv = tid >> 5;
    int half = lane >> 4, N = lane & 15;

    // Stage A = [x_t | h_{t-1}] : 16 rows x 1024 bf16
    for (int i = tid; i < 16 * 1024; i += blockDim.x) {
        int r = i >> 10, k = i & 1023;
        int row = r0 + r;
        sA[i] = (k < 512) ? steps[(size_t)row * PLY * EMB + (size_t)t * EMB + k]
                          : hbf[(size_t)row * EMB + (k - 512)];
    }
    __syncthreads();

    // gates = A @ [W_ih;W_hh]^T (+bias): each wave owns 4 e-tiles, i/f/g/o each
#pragma unroll
    for (int ii = 0; ii < 4; ++ii) {
        int et = wv + 8 * ii;                      // 0..31 e-tiles of 16
        v8f ai = {}, af = {}, ag = {}, ao = {};
        for (int kc = 0; kc < 1024; kc += 32) {
            v16bf a = load_a16(sA, 1024, lane, kc);
            ai = wmma_bf16(a, load_b16(Wcat, 1024, lane,    0 + et * 16, kc), ai);
            af = wmma_bf16(a, load_b16(Wcat, 1024, lane,  512 + et * 16, kc), af);
            ag = wmma_bf16(a, load_b16(Wcat, 1024, lane, 1024 + et * 16, kc), ag);
            ao = wmma_bf16(a, load_b16(Wcat, 1024, lane, 1536 + et * 16, kc), ao);
        }
        int e = et * 16 + N;
        float bi = bcomb[e], bf_ = bcomb[512 + e], bg = bcomb[1024 + e], bo = bcomb[1536 + e];
#pragma unroll
        for (int k = 0; k < 8; ++k) {              // D layout: M = k + 8*(lane/16)
            int row = r0 + k + 8 * half;
            float gi = ai[k] + bi, gf = af[k] + bf_, gg = ag[k] + bg, go = ao[k] + bo;
            float si = 1.f / (1.f + __expf(-gi));
            float sf = 1.f / (1.f + __expf(-gf));
            float so = 1.f / (1.f + __expf(-go));
            float cn = sf * cbuf[(size_t)row * EMB + e] + si * tanhf(gg);
            float hn = so * tanhf(cn);
            cbuf[(size_t)row * EMB + e] = cn;
            bf16s hb = f2bf(hn);
            hbf[(size_t)row * EMB + e] = hb;       // feeds next step's A stage
            sH[(k + 8 * half) * 512 + e] = hb;     // feeds q GEMM below
        }
    }
    __syncthreads();

    // q = h_new @ Wq^T + bq  (K=512) -> f32 q-buffer overlaid on sA
    float* qbuf = reinterpret_cast<float*>(sA);
    {
        v8f acc[4] = {};
        for (int kc = 0; kc < 512; kc += 32) {
            v16bf a = load_a16(sH, 512, lane, kc);
#pragma unroll
            for (int ii = 0; ii < 4; ++ii)
                acc[ii] = wmma_bf16(a, load_b16(Wqb, 512, lane, (wv + 8 * ii) * 16, kc), acc[ii]);
        }
#pragma unroll
        for (int ii = 0; ii < 4; ++ii) {
            int col = (wv + 8 * ii) * 16 + N;
            float bqv = bq[col];
#pragma unroll
            for (int k = 0; k < 8; ++k)
                qbuf[(k + 8 * half) * 512 + col] = acc[ii][k] + bqv;
        }
    }
    __syncthreads();

    // scores sc[r][p] = sum_e v[e]*tanh(q[r][e] + ctx[row,p,e])  (ctx is L2-hot)
    for (int pr = tid; pr < 16 * 32; pr += blockDim.x) {
        int r = pr >> 5, p = pr & 31, row = r0 + r;
        const float* ctxr = ctx + ((size_t)row * PLY + p) * EMB;
        const float* qr = qbuf + r * 512;
        float s = 0.f;
        for (int e = 0; e < EMB; ++e) s += vvec[e] * tanhf(qr[e] + ctxr[e]);
        sSc[pr] = s;
    }
    __syncthreads();

    if (tid < 16) {                                // per-row log-softmax stats + argmax
        const float* sc = sSc + tid * 32;
        float m = sc[0]; int am = 0;
        for (int p = 1; p < 32; ++p) if (sc[p] > m) { m = sc[p]; am = p; }
        float se = 0.f;
        for (int p = 0; p < 32; ++p) se += __expf(sc[p] - m);
        sMl[tid] = m + __logf(se);
        out_arg[(size_t)(r0 + tid) * PLY + t] = (float)am;
    }
    __syncthreads();

    for (int pr = tid; pr < 16 * 32; pr += blockDim.x) {
        int r = pr >> 5, p = pr & 31;
        out_att[((size_t)(r0 + r) * PLY + t) * PLY + p] = sSc[pr] - sMl[r];
    }
}

// ---- launcher --------------------------------------------------------------
extern "C" void kernel_launch(void* const* d_in, const int* in_sizes, int n_in,
                              void* d_out, int out_size, void* d_ws, size_t ws_size,
                              hipStream_t stream) {
    const float* enc[3] = {(const float*)d_in[0], (const float*)d_in[2], (const float*)d_in[4]};
    const int*   idx[3] = {(const int*)d_in[1], (const int*)d_in[3], (const int*)d_in[5]};
    const float* init_embed = (const float*)d_in[6];
    const float* W_ih = (const float*)d_in[7];
    const float* W_hh = (const float*)d_in[8];
    const float* b_ih = (const float*)d_in[9];
    const float* b_hh = (const float*)d_in[10];
    const float* Wq   = (const float*)d_in[11];
    const float* bq   = (const float*)d_in[12];
    const float* Wc   = (const float*)d_in[13];
    const float* v    = (const float*)d_in[14];
    float* out = (float*)d_out;

    char* ws = (char*)d_ws;
    size_t off = 0;
    auto alloc = [&](size_t bytes) -> void* {
        void* p = ws + off;
        off += (bytes + 255) & ~(size_t)255;
        return p;
    };
    bf16s* Wcat  = (bf16s*)alloc((size_t)2048 * 1024 * 2);
    bf16s* Wqb   = (bf16s*)alloc((size_t)512 * 512 * 2);
    bf16s* Wcb   = (bf16s*)alloc((size_t)512 * 512 * 2);
    float* bcomb = (float*)alloc((size_t)2048 * 4);
    bf16s* steps = (bf16s*)alloc((size_t)BTOT * PLY * EMB * 2);
    float* ctx   = (float*)alloc((size_t)BTOT * PLY * EMB * 4);
    float* cbuf  = (float*)alloc((size_t)BTOT * EMB * 4);
    bf16s* hbf   = (bf16s*)alloc((size_t)BTOT * EMB * 2);

    k_prep<<<4096, 256, 0, stream>>>(W_ih, W_hh, b_ih, b_hh, Wq, Wc, Wcat, Wqb, Wcb, bcomb);

    for (int tau = 0; tau < 3; ++tau) {
        k_gather<<<BTOT, 256, 0, stream>>>(enc[tau], idx[tau], init_embed, steps, cbuf, hbf);
        k_ctx<<<(BTOT * PLY) / 16, 256, 0, stream>>>(enc[tau], Wcb, ctx);
        float* oatt = out + (size_t)tau * BTOT * PLY * PLY;
        float* oarg = out + (size_t)3 * BTOT * PLY * PLY + (size_t)tau * BTOT * PLY;
        for (int t = 0; t < PLY; ++t)
            k_step<<<BTOT / 16, 256, 0, stream>>>(t, steps, Wcat, bcomb, Wqb, bq, ctx, v,
                                                  cbuf, hbf, oatt, oarg);
    }
}
